// L3PANConv_84859963834443
// MI455X (gfx1250) — matrix operand order repaired and checked by reading, not compiled, not run on hardware
//
#include <hip/hip_runtime.h>
#include <hip/hip_bf16.h>

// ---------------------------------------------------------------------------
// PANConv x3 for MI455X (gfx1250, wave32, WMMA f16 -> f32 accumulate)
//
//   A, A^T         : dense 0/1 adjacency from edge_index (scatter)
//   S5 support     : R0 = support(I+A); 4x boolean GEMM  S_{k+1}=supp(S_k+S_k A)
//   d              : rowsum(S5)^-1/2   (exact: 0/1 sums in f32 acc)
//   layer1: y0=d*x; y_i=A y_{i-1}, z=sum c_i y_i; v=d*z; h1=relu(v@W1+b1)
//   layer2: u=h1@W2 (epilogue: y0=d*u, z=c0*y0); 5x propagate; h2=relu(d*z+b2)
//   layer3: same with W3; out=relu(d*z+b3) -> f32 d_out
//
// GEMM: wave tile (16*MT)x(16*NT); K loop unrolled 2x with ping-pong register
// buffers (no copy moves), loads of one phase overlap WMMAs of the other.
// Requires K % 64 == 0 (true for 2048 / 3200 / 1600 / 128).
// ---------------------------------------------------------------------------

#define NNODES 2048
#define NEDGES 65536
#define CH_IN  128
#define CH_H1  3200
#define CH_H2  1600
#define CH_OUT 32

typedef _Float16 h8   __attribute__((ext_vector_type(8)));
typedef _Float16 v16h __attribute__((ext_vector_type(16)));
typedef float    v8f  __attribute__((ext_vector_type(8)));

enum { EPI_BOOL = 0, EPI_PROPINIT = 1, EPI_PROPAGATE = 2, EPI_BIASRELU = 3 };

// Operand layouts per CDNA5 ISA 7.12.2 (wave32):
//   A: lane(0..15)=row, halves[0..7]=K(kb+8*kh .. +7), halves[8..15]=K(kb+16+8*kh ..)
//   B: lane(0..15)=col, halves[0..7]=K(kb+8*kh .. +7), halves[8..15]=K(kb+16+8*kh ..)
//   C/D: col = n&15 per lane, rows = m0 + 8*(lane>>4) + r for vgpr r
template <int EPI, int MT, int NT>
__global__ __launch_bounds__(256) void gemm16(
    const _Float16* __restrict__ Am, const _Float16* __restrict__ BT,
    int M, int Nn, int K,
    const float* __restrict__ dvec,   // PROPINIT
    const float* __restrict__ wvec,   // PROPINIT/PROPAGATE: filter weights
    int widx,                         // PROPAGATE: power index i (c = w0*...*wi)
    const float* __restrict__ bias,   // BIASRELU
    const _Float16* __restrict__ Rold,// BOOL
    _Float16* __restrict__ outH,      // f16 output (row-major or transposed)
    float* __restrict__ zT)           // PROPINIT/PROPAGATE: z accumulator [Nn][M]
{
  const int wave    = (int)((blockIdx.x * blockDim.x + threadIdx.x) >> 5);
  const int lane    = (int)(threadIdx.x & 31);
  const int nGroups = Nn / (16 * NT);
  const int nTiles  = (M / (16 * MT)) * nGroups;
  if (wave >= nTiles) return;
  const int m0 = (wave / nGroups) * (16 * MT);
  const int n0 = (wave % nGroups) * (16 * NT);
  const int l  = lane & 15;
  const int kh = lane >> 4;

  const _Float16* arow[MT];
#pragma unroll
  for (int t = 0; t < MT; ++t) arow[t] = Am + (size_t)(m0 + t * 16 + l) * K;
  const _Float16* brow[NT];
#pragma unroll
  for (int j = 0; j < NT; ++j) brow[j] = BT + (size_t)(n0 + j * 16 + l) * K;

  auto ldA = [&](int t, int k) -> v16h {
    h8 lo = *(const h8*)(arow[t] + k + kh * 8);
    h8 hi = *(const h8*)(arow[t] + k + 16 + kh * 8);
    return __builtin_shufflevector(lo, hi, 0, 1, 2, 3, 4, 5, 6, 7,
                                   8, 9, 10, 11, 12, 13, 14, 15);
  };
  auto ldB = [&](int j, int k) -> v16h {
    h8 lo = *(const h8*)(brow[j] + k + kh * 8);
    h8 hi = *(const h8*)(brow[j] + k + 16 + kh * 8);
    return __builtin_shufflevector(lo, hi, 0, 1, 2, 3, 4, 5, 6, 7,
                                   8, 9, 10, 11, 12, 13, 14, 15);
  };

  v8f acc[MT][NT] = {};
  v16h a0[MT], b0[NT], a1[MT], b1[NT];

  auto loadAll = [&](v16h* av, v16h* bv, int k) {
#pragma unroll
    for (int t = 0; t < MT; ++t) av[t] = ldA(t, k);
#pragma unroll
    for (int j = 0; j < NT; ++j) bv[j] = ldB(j, k);
  };
  auto mulAll = [&](const v16h* av, const v16h* bv) {
#pragma unroll
    for (int t = 0; t < MT; ++t)
#pragma unroll
      for (int j = 0; j < NT; ++j)
        acc[t][j] = __builtin_amdgcn_wmma_f32_16x16x32_f16(
            false, av[t], false, bv[j], (short)0, acc[t][j], false, false);
  };

  // ping-pong pipelined K loop (K % 64 == 0, K >= 64)
  loadAll(a0, b0, 0);
  for (int k = 32; k + 32 < K; k += 64) {
    loadAll(a1, b1, k);        // phase-1 loads overlap phase-0 WMMAs
    mulAll(a0, b0);
    loadAll(a0, b0, k + 32);   // phase-0 loads overlap phase-1 WMMAs
    mulAll(a1, b1);
  }
  loadAll(a1, b1, K - 32);
  mulAll(a0, b0);
  mulAll(a1, b1);

  // -------------------------- epilogues ------------------------------------
  float c = 0.0f;
  if (EPI == EPI_PROPINIT || EPI == EPI_PROPAGATE) {
    c = wvec[0];
    if (EPI == EPI_PROPAGATE)
      for (int q = 1; q <= widx; ++q) c *= wvec[q];
  }
#pragma unroll
  for (int t = 0; t < MT; ++t) {
    const int rowb = m0 + t * 16 + kh * 8;
    if (EPI == EPI_BOOL) {
      // Rnew = (acc + Rold) > 0  (counts exact in f32; entries 0/1)
#pragma unroll
      for (int j = 0; j < NT; ++j) {
        const int col = n0 + j * 16 + l;
#pragma unroll
        for (int r = 0; r < 8; ++r) {
          size_t idx = (size_t)(rowb + r) * Nn + col;
          float v = acc[t][j][r] + (float)Rold[idx];
          outH[idx] = (v > 0.5f) ? (_Float16)1.0f : (_Float16)0.0f;
        }
      }
    } else if (EPI == EPI_BIASRELU) {
#pragma unroll
      for (int j = 0; j < NT; ++j) {
        const int col = n0 + j * 16 + l;
        const float bb = bias[col];
#pragma unroll
        for (int r = 0; r < 8; ++r) {
          float v = fmaxf(acc[t][j][r] + bb, 0.0f);
          outH[(size_t)(rowb + r) * Nn + col] = (_Float16)v;
        }
      }
    } else {  // PROPINIT / PROPAGATE: write yT[col][row] (contig) + z accum
#pragma unroll
      for (int j = 0; j < NT; ++j) {
        const int col = n0 + j * 16 + l;
        const size_t base = (size_t)col * M + rowb;
        h8 y;
#pragma unroll
        for (int r = 0; r < 8; ++r) {
          float s = acc[t][j][r];
          if (EPI == EPI_PROPINIT) {
            s *= dvec[rowb + r];      // fold d-scaling of M_hat input
            zT[base + r] = c * s;     // z = c0 * y0
          } else {
            zT[base + r] += c * s;    // z += c_i * y_i
          }
          y[r] = (_Float16)s;
        }
        *(h8*)(outH + base) = y;      // one 16B store (transposed layout)
      }
    }
  }
}

// ---------------------------------------------------------------------------
// Small helper kernels
// ---------------------------------------------------------------------------
__global__ __launch_bounds__(256) void scatter_edges(
    const int* __restrict__ ei, _Float16* __restrict__ A, _Float16* __restrict__ AT) {
  int e = blockIdx.x * 256 + threadIdx.x;
  if (e >= NEDGES) return;
  int s = ei[e];             // edge_index[0][e] = src
  int d = ei[NEDGES + e];    // edge_index[1][e] = dst
  A[(size_t)d * NNODES + s]  = (_Float16)1.0f;  // A[dst,src] = 1
  AT[(size_t)s * NNODES + d] = (_Float16)1.0f;
}

__global__ __launch_bounds__(256) void init_reach(
    const _Float16* __restrict__ A, _Float16* __restrict__ R) {
  size_t idx = (size_t)blockIdx.x * 256 + threadIdx.x;
  int m = (int)(idx >> 11), n = (int)(idx & (NNODES - 1));
  float a = (float)A[idx];
  R[idx] = (a != 0.0f || m == n) ? (_Float16)1.0f : (_Float16)0.0f;
}

__global__ __launch_bounds__(256) void rowcount_rsqrt(
    const _Float16* __restrict__ R, float* __restrict__ dvec) {
  __shared__ float sm[256];
  int row = blockIdx.x;
  float s = 0.0f;
  for (int j = threadIdx.x; j < NNODES; j += 256)
    s += (float)R[(size_t)row * NNODES + j];
  sm[threadIdx.x] = s;
  __syncthreads();
  for (int off = 128; off > 0; off >>= 1) {
    if ((int)threadIdx.x < off) sm[threadIdx.x] += sm[threadIdx.x + off];
    __syncthreads();
  }
  if (threadIdx.x == 0) {
    float deg = sm[0];
    dvec[row] = (deg > 0.0f) ? rsqrtf(deg) : 0.0f;
  }
}

__global__ __launch_bounds__(256) void transpose_w(
    const float* __restrict__ W, _Float16* __restrict__ WT, int K, int Nn) {
  size_t idx = (size_t)blockIdx.x * 256 + threadIdx.x;
  if (idx >= (size_t)K * Nn) return;
  int n = (int)(idx % Nn), k = (int)(idx / Nn);
  WT[(size_t)n * K + k] = (_Float16)W[idx];
}

// layer-1 entry: y0T = d*x (transposed), zT = w0*y0
__global__ __launch_bounds__(256) void scale_init_x(
    const float* __restrict__ x, const float* __restrict__ dvec,
    const float* __restrict__ w, _Float16* __restrict__ y0T,
    float* __restrict__ zT) {
  int idx = blockIdx.x * 256 + threadIdx.x;  // 2048*128
  int n = idx & (CH_IN - 1), m = idx >> 7;
  float s = dvec[m] * x[(size_t)m * CH_IN + n];
  size_t t = (size_t)n * NNODES + m;
  y0T[t] = (_Float16)s;
  zT[t]  = w[0] * s;
}

enum { FIN_F16_PLAIN = 0, FIN_F16_BIAS_RELU = 1, FIN_F32_BIAS_RELU = 2 };

template <int MODE>
__global__ __launch_bounds__(256) void finalize(
    const float* __restrict__ zT, const float* __restrict__ dvec,
    const float* __restrict__ bias, _Float16* __restrict__ outH,
    float* __restrict__ outF, int Nn) {
  size_t idx = (size_t)blockIdx.x * 256 + threadIdx.x;  // Nn * 2048, n-major
  if (idx >= (size_t)Nn * NNODES) return;
  int m = (int)(idx & (NNODES - 1)), n = (int)(idx >> 11);
  float v = dvec[m] * zT[(size_t)n * NNODES + m];
  if (MODE != FIN_F16_PLAIN) v = fmaxf(v + bias[n], 0.0f);
  if (MODE == FIN_F32_BIAS_RELU) outF[(size_t)m * Nn + n] = v;
  else                           outH[(size_t)m * Nn + n] = (_Float16)v;
}

// ---------------------------------------------------------------------------
static inline int gemm_blocks(int M, int Nn, int MT, int NT) {
  return ((M / (16 * MT)) * (Nn / (16 * NT)) * 32 + 255) / 256;
}

extern "C" void kernel_launch(void* const* d_in, const int* in_sizes, int n_in,
                              void* d_out, int out_size, void* d_ws, size_t ws_size,
                              hipStream_t stream) {
  (void)in_sizes; (void)n_in; (void)out_size; (void)ws_size;
  const float* x  = (const float*)d_in[0];
  const int*   ei = (const int*)d_in[1];
  const float* w1 = (const float*)d_in[2];
  const float* w2 = (const float*)d_in[3];
  const float* w3 = (const float*)d_in[4];
  const float* W1 = (const float*)d_in[5];
  const float* b1 = (const float*)d_in[6];
  const float* W2 = (const float*)d_in[7];
  const float* b2 = (const float*)d_in[8];
  const float* W3 = (const float*)d_in[9];
  const float* b3 = (const float*)d_in[10];
  float* out = (float*)d_out;

  // workspace carve-out (~116 MB)
  char* p = (char*)d_ws;
  auto carve = [&](size_t bytes) {
    char* r = p;
    p += (bytes + 255) & ~(size_t)255;
    return r;
  };
  const size_t NN = (size_t)NNODES * NNODES;
  _Float16* A    = (_Float16*)carve(NN * 2);
  _Float16* ATr  = (_Float16*)carve(NN * 2);
  _Float16* R0   = (_Float16*)carve(NN * 2);
  _Float16* R1   = (_Float16*)carve(NN * 2);
  float*    dvec = (float*)carve(NNODES * 4);
  _Float16* W1T  = (_Float16*)carve((size_t)CH_H1 * CH_IN * 2);
  _Float16* W2T  = (_Float16*)carve((size_t)CH_H2 * CH_H1 * 2);
  _Float16* W3T  = (_Float16*)carve((size_t)CH_OUT * CH_H2 * 2);
  _Float16* yTa  = (_Float16*)carve((size_t)CH_H1 * NNODES * 2);
  _Float16* yTb  = (_Float16*)carve((size_t)CH_H1 * NNODES * 2);
  float*    zT   = (float*)carve((size_t)CH_H1 * NNODES * 4);
  _Float16* vbuf = (_Float16*)carve((size_t)NNODES * CH_IN * 2);
  _Float16* h1   = (_Float16*)carve((size_t)NNODES * CH_H1 * 2);
  _Float16* h2   = (_Float16*)carve((size_t)NNODES * CH_H2 * 2);

  // 1) adjacency
  hipMemsetAsync(A, 0, NN * 2, stream);
  hipMemsetAsync(ATr, 0, NN * 2, stream);
  scatter_edges<<<NEDGES / 256, 256, 0, stream>>>(ei, A, ATr);

  // 2) reach support S5 and d = deg^-1/2  (4 boolean WMMA GEMMs)
  init_reach<<<(int)(NN / 256), 256, 0, stream>>>(A, R0);
  _Float16* Rs = R0;
  _Float16* Rd = R1;
  for (int it = 0; it < 4; ++it) {
    gemm16<EPI_BOOL, 2, 4><<<gemm_blocks(NNODES, NNODES, 2, 4), 256, 0, stream>>>(
        Rs, ATr, NNODES, NNODES, NNODES, nullptr, nullptr, 0, nullptr, Rs, Rd,
        nullptr);
    _Float16* t = Rs; Rs = Rd; Rd = t;
  }
  rowcount_rsqrt<<<NNODES, 256, 0, stream>>>(Rs, dvec);

  // 3) weight transposes (B-operand layout)
  transpose_w<<<(CH_IN * CH_H1) / 256, 256, 0, stream>>>(W1, W1T, CH_IN, CH_H1);
  transpose_w<<<(CH_H1 * CH_H2) / 256, 256, 0, stream>>>(W2, W2T, CH_H1, CH_H2);
  transpose_w<<<(CH_H2 * CH_OUT) / 256, 256, 0, stream>>>(W3, W3T, CH_H2, CH_OUT);

  // 4) layer 1: propagate at C=128, then W1
  scale_init_x<<<(NNODES * CH_IN) / 256, 256, 0, stream>>>(x, dvec, w1, yTa, zT);
  {
    _Float16* yi = yTa; _Float16* yo = yTb;
    for (int i = 1; i <= 5; ++i) {
      gemm16<EPI_PROPAGATE, 2, 4>
          <<<gemm_blocks(NNODES, CH_IN, 2, 4), 256, 0, stream>>>(
              A, yi, NNODES, CH_IN, NNODES, nullptr, w1, i, nullptr, nullptr,
              yo, zT);
      _Float16* t = yi; yi = yo; yo = t;
    }
  }
  finalize<FIN_F16_PLAIN><<<(NNODES * CH_IN) / 256, 256, 0, stream>>>(
      zT, dvec, nullptr, vbuf, nullptr, CH_IN);
  gemm16<EPI_BIASRELU, 2, 4><<<gemm_blocks(NNODES, CH_H1, 2, 4), 256, 0, stream>>>(
      vbuf, W1T, NNODES, CH_H1, CH_IN, nullptr, nullptr, 0, b1, nullptr, h1,
      nullptr);

  // 5) layer 2: W2 first (3200->1600, epilogue starts propagation), 5x A
  gemm16<EPI_PROPINIT, 2, 4><<<gemm_blocks(NNODES, CH_H2, 2, 4), 256, 0, stream>>>(
      h1, W2T, NNODES, CH_H2, CH_H1, dvec, w2, 0, nullptr, nullptr, yTa, zT);
  {
    _Float16* yi = yTa; _Float16* yo = yTb;
    for (int i = 1; i <= 5; ++i) {
      gemm16<EPI_PROPAGATE, 2, 4>
          <<<gemm_blocks(NNODES, CH_H2, 2, 4), 256, 0, stream>>>(
              A, yi, NNODES, CH_H2, NNODES, nullptr, w2, i, nullptr, nullptr,
              yo, zT);
      _Float16* t = yi; yi = yo; yo = t;
    }
  }
  finalize<FIN_F16_BIAS_RELU><<<(NNODES * CH_H2) / 256, 256, 0, stream>>>(
      zT, dvec, b2, h2, nullptr, CH_H2);

  // 6) layer 3: W3 first (1600->32), 5x A, f32 output
  gemm16<EPI_PROPINIT, 1, 2><<<gemm_blocks(NNODES, CH_OUT, 1, 2), 256, 0, stream>>>(
      h2, W3T, NNODES, CH_OUT, CH_H2, dvec, w3, 0, nullptr, nullptr, yTa, zT);
  {
    _Float16* yi = yTa; _Float16* yo = yTb;
    for (int i = 1; i <= 5; ++i) {
      gemm16<EPI_PROPAGATE, 1, 2>
          <<<gemm_blocks(NNODES, CH_OUT, 1, 2), 256, 0, stream>>>(
              A, yi, NNODES, CH_OUT, NNODES, nullptr, w3, i, nullptr, nullptr,
              yo, zT);
      _Float16* t = yi; yi = yo; yo = t;
    }
  }
  finalize<FIN_F32_BIAS_RELU><<<(NNODES * CH_OUT) / 256, 256, 0, stream>>>(
      zT, dvec, b3, nullptr, out, CH_OUT);
}
